// myLSTM_21139829030986
// MI455X (gfx1250) — matrix operand order repaired and checked by reading, not compiled, not run on hardware
//
#include <hip/hip_runtime.h>
#include <hip/hip_bf16.h>

// ---------------- types ----------------
typedef __bf16 bf16_t;
typedef __attribute__((ext_vector_type(16))) __bf16 bf16x16;
typedef __attribute__((ext_vector_type(8)))  float  f32x8;

union Frag { uint4 u[2]; bf16x16 v; };

#define T_STEPS 256
#define BATCH   64
#define INSZ    1024
#define HSZ     1024
#define G4H     4096   // 4*H

#define LDSTRIDE 40    // padded LDS row stride in bf16 elems (80 B = 20 dwords)

__device__ __forceinline__ float sigmoidf_fast(float x) {
    return 1.0f / (1.0f + __expf(-x));
}

// ---------------- CDNA5 async global->LDS (ASYNCcnt-tracked, no VGPR staging) ----
// global_load_async_to_lds_b128: vdst = LDS byte address, vaddr = 64-bit VA.
__device__ __forceinline__ void async_ld_b128(unsigned lds_addr, const void* gaddr) {
    asm volatile("global_load_async_to_lds_b128 %0, %1, off"
                 :: "v"(lds_addr), "v"(gaddr) : "memory");
}
__device__ __forceinline__ void async_ld_b128x2(unsigned lds_addr, const bf16_t* gaddr) {
    asm volatile("global_load_async_to_lds_b128 %0, %2, off\n\t"
                 "global_load_async_to_lds_b128 %1, %3, off"
                 :: "v"(lds_addr), "v"(lds_addr + 16u), "v"(gaddr), "v"(gaddr + 8)
                 : "memory");
}
#define WAIT_ASYNCCNT(n) asm volatile("s_wait_asynccnt %0" :: "i"(n) : "memory")

// ---------------- f32 -> bf16 convert (8 elems / thread) ----------------
__launch_bounds__(256)
__global__ void f32_to_bf16_kernel(const float* __restrict__ in,
                                   bf16_t* __restrict__ out, int n) {
    int i = (blockIdx.x * blockDim.x + threadIdx.x) * 8;
    if (i >= n) return;
    const float4 a = *(const float4*)(in + i);
    const float4 b = *(const float4*)(in + i + 4);
    union { bf16_t h[8]; uint4 u; } r;
    r.h[0] = (bf16_t)a.x; r.h[1] = (bf16_t)a.y;
    r.h[2] = (bf16_t)a.z; r.h[3] = (bf16_t)a.w;
    r.h[4] = (bf16_t)b.x; r.h[5] = (bf16_t)b.y;
    r.h[6] = (bf16_t)b.z; r.h[7] = (bf16_t)b.w;
    *(uint4*)(out + i) = r.u;
}

// ---------------- fragment builders (contiguous K order, same for A and B) ----------------
// A 16x32 bf16: lanes 0-15 -> M=lane, K 0..15; lanes 16-31 -> M=lane-16, K 16..31
__device__ __forceinline__ bf16x16 load_a_frag(const bf16_t* tile, int mBase, int lane) {
    Frag f;
    const int m  = mBase + (lane & 15);
    const int hk = (lane >> 4) * 16;
    const bf16_t* p = tile + m * LDSTRIDE + hk;
    f.u[0] = *(const uint4*)(p);        // K = hk .. hk+7
    f.u[1] = *(const uint4*)(p + 8);    // K = hk+8 .. hk+15
    return f.v;
}
// B 32x16 bf16: lanes 0-15 -> N=lane, K 0..15; lanes 16-31 -> N=lane-16, K 16..31
// tile stored as B^T[n][k] (== weight-row-major), so per lane: 16 contiguous K.
__device__ __forceinline__ bf16x16 load_b_frag(const bf16_t* tile, int nBase, int lane) {
    Frag f;
    const int n  = nBase + (lane & 15);
    const int kb = (lane >> 4) * 16;
    const bf16_t* p = tile + n * LDSTRIDE + kb;
    f.u[0] = *(const uint4*)(p);
    f.u[1] = *(const uint4*)(p + 8);
    return f.v;
}

// ---------------- GEMM1: xproj[M=16384][N=4096] = Xb * Wi^T + (bi+bh) ----------------
// WG tile 128x128, 8 waves (2 M x 4 N), wave tile 64x32 = 4x2 WMMA tiles, K-step 32.
// Dynamic LDS layout (byte offsets from 0):
//   buf0: A @ 0       (128*80 = 10240)   B @ 10240 (10240)
//   buf1: A @ 20480                      B @ 30720        -> total 40960 B
#define G1_BUFSZ 20480u
__launch_bounds__(256)
__global__ void gemm_xproj_kernel(const bf16_t* __restrict__ Xb,     // [16384][1024]
                                  const bf16_t* __restrict__ Wib,    // [4096][1024]
                                  const float*  __restrict__ bi,
                                  const float*  __restrict__ bh,
                                  float* __restrict__ xproj) {       // [16384][4096]
    extern __shared__ char smem[];

    const int m0   = blockIdx.y * 128;
    const int n0   = blockIdx.x * 128;
    const int tid  = threadIdx.x;
    const int lane = tid & 31;
    const int wid  = tid >> 5;
    const int wm   = (wid & 1) * 64;   // wave M offset inside WG tile
    const int wn   = (wid >> 1) * 32;  // wave N offset inside WG tile

    // cooperative tile fetch: 128x32 bf16 per matrix, 32 B (2 x b128) per thread each
    const int lrow = tid >> 1;
    const int lk   = (tid & 1) * 16;
    const bf16_t* gA = Xb  + (size_t)(m0 + lrow) * INSZ + lk;   // advance by k0
    const bf16_t* gB = Wib + (size_t)(n0 + lrow) * INSZ + lk;
    const unsigned ldsOff = (unsigned)(lrow * (LDSTRIDE * 2) + lk * 2); // bytes in buffer

    // prologue: tile 0 -> buffer 0 (4 async instructions / wave)
    async_ld_b128x2(ldsOff,          gA);
    async_ld_b128x2(10240u + ldsOff, gB);

    f32x8 acc[4][2] = {};

    for (int k0 = 0; k0 < INSZ; k0 += 32) {
        const int cur = (k0 >> 5) & 1;
        if (k0 + 32 < INSZ) {   // overlap: fetch tile k+1 into the other buffer
            const unsigned nb = (unsigned)(1 - cur) * G1_BUFSZ;
            async_ld_b128x2(nb + ldsOff,          gA + k0 + 32);
            async_ld_b128x2(nb + 10240u + ldsOff, gB + k0 + 32);
            WAIT_ASYNCCNT(4);   // current tile's 4 loads retired (in-order)
        } else {
            WAIT_ASYNCCNT(0);
        }
        __syncthreads();

        const bf16_t* As = (const bf16_t*)(smem + cur * G1_BUFSZ);
        const bf16_t* Bs = (const bf16_t*)(smem + cur * G1_BUFSZ + 10240u);

        bf16x16 af[4], bfr[2];
#pragma unroll
        for (int mt = 0; mt < 4; ++mt) af[mt] = load_a_frag(As, wm + mt * 16, lane);
#pragma unroll
        for (int nt = 0; nt < 2; ++nt) bfr[nt] = load_b_frag(Bs, wn + nt * 16, lane);
#pragma unroll
        for (int mt = 0; mt < 4; ++mt)
#pragma unroll
            for (int nt = 0; nt < 2; ++nt)
                acc[mt][nt] = __builtin_amdgcn_wmma_f32_16x16x32_bf16(
                    false, af[mt], false, bfr[nt], (short)0, acc[mt][nt], false, false);
        __syncthreads();        // everyone done reading buf[cur] before it is refilled
    }

    // epilogue: fuse (bi+bh), store f32
#pragma unroll
    for (int mt = 0; mt < 4; ++mt)
#pragma unroll
        for (int nt = 0; nt < 2; ++nt) {
            const int mb = m0 + wm + mt * 16 + ((lane >> 4) << 3);
            const int n  = n0 + wn + nt * 16 + (lane & 15);
            const float bsum = bi[n] + bh[n];
#pragma unroll
            for (int r = 0; r < 8; ++r)
                xproj[(size_t)(mb + r) * G4H + n] = acc[mt][nt][r] + bsum;
        }
}

// ---------------- recurrent step: gates = xproj[t] + h @ Wh^T; LSTM cell update ----------
// 32 WGs; WG `wg` owns H columns [wg*32, wg*32+32) and all 4 gate blocks for them
// (GEMM N-cols c in [0,128): gate g = c/32, j = c%32 -> Wh row g*1024 + wg*32 + j).
// Dynamic LDS layout (byte offsets):
//   buf0: Ah @ 0     (64*80 = 5120)   Bh @ 5120  (128*80 = 10240)
//   buf1: Ah @ 15360                  Bh @ 20480
//   gbuf  @ 30720  (64 * 132 * 4 = 33792)        -> total 64512 B
#define ST_BUFSZ 15360u
#define ST_GOFF  30720u
__launch_bounds__(256)
__global__ void lstm_step_kernel(int t,
                                 const float*  __restrict__ xproj,   // [T*B][4096]
                                 const bf16_t* __restrict__ Whb,     // [4096][1024]
                                 const bf16_t* __restrict__ h_in,    // [64][1024] bf16
                                 bf16_t*       __restrict__ h_out,   // [64][1024] bf16
                                 float*        __restrict__ cstate,  // [64][1024] f32
                                 float*        __restrict__ out) {   // [T][64][1024]
    extern __shared__ char smem[];
    float* gbuf = (float*)(smem + ST_GOFF);     // gates f32, padded stride 132

    const int wg   = blockIdx.x;
    const int tid  = threadIdx.x;
    const int lane = tid & 31;
    const int wid  = tid >> 5;          // wave owns N-cols [wid*16, wid*16+16)

    f32x8 acc[4] = {};                  // 4 M-tiles (M = 64 batch)

    // loader mappings (async, no VGPR staging): Ah 1 x b128/thread, Bh 2 x b128/thread
    const int ar = tid >> 2, ak = (tid & 3) * 8;           // Ah: 64x32
    const int bc = tid >> 1, bk = (tid & 1) * 16;          // Bh: 128x32
    const int wr = (bc >> 5) * HSZ + wg * 32 + (bc & 31);  // Wh row for local col bc
    const bf16_t* gH = h_in + (size_t)ar * HSZ + ak;       // advance by k0
    const bf16_t* gW = Whb  + (size_t)wr * HSZ + bk;
    const unsigned ldsA = (unsigned)(ar * (LDSTRIDE * 2) + ak * 2);
    const unsigned ldsB = 5120u + (unsigned)(bc * (LDSTRIDE * 2) + bk * 2);

    // prologue: tile 0 -> buffer 0 (3 async instructions / wave)
    async_ld_b128(ldsA, gH);
    async_ld_b128x2(ldsB, gW);

    for (int k0 = 0; k0 < HSZ; k0 += 32) {
        const int cur = (k0 >> 5) & 1;
        if (k0 + 32 < HSZ) {
            const unsigned nb = (unsigned)(1 - cur) * ST_BUFSZ;
            async_ld_b128(nb + ldsA, gH + k0 + 32);
            async_ld_b128x2(nb + ldsB, gW + k0 + 32);
            WAIT_ASYNCCNT(3);
        } else {
            WAIT_ASYNCCNT(0);
        }
        __syncthreads();

        const bf16_t* Ah = (const bf16_t*)(smem + cur * ST_BUFSZ);
        const bf16_t* Bh = (const bf16_t*)(smem + cur * ST_BUFSZ + 5120u);

        const bf16x16 bfr = load_b_frag(Bh, wid * 16, lane);
#pragma unroll
        for (int mt = 0; mt < 4; ++mt) {
            const bf16x16 af = load_a_frag(Ah, mt * 16, lane);
            acc[mt] = __builtin_amdgcn_wmma_f32_16x16x32_bf16(
                false, af, false, bfr, (short)0, acc[mt], false, false);
        }
        __syncthreads();
    }

    // spill accumulators to LDS so each thread can gather all 4 gates of a column
#pragma unroll
    for (int mt = 0; mt < 4; ++mt) {
        const int mb = mt * 16 + ((lane >> 4) << 3);
        const int nc = wid * 16 + (lane & 15);
#pragma unroll
        for (int r = 0; r < 8; ++r)
            gbuf[(mb + r) * 132 + nc] = acc[mt][r];
    }
    __syncthreads();

    // fused gate nonlinearity + cell/hidden update: 64x32 outputs, 8 per thread
#pragma unroll
    for (int i = 0; i < 8; ++i) {
        const int e = tid * 8 + i;
        const int b = e >> 5;
        const int j = e & 31;
        const size_t xrow = ((size_t)t * BATCH + b) * G4H + wg * 32 + j;
        const float gi = sigmoidf_fast(xproj[xrow          ] + gbuf[b * 132 +      j]);
        const float gf = sigmoidf_fast(xproj[xrow +     HSZ] + gbuf[b * 132 + 32 + j]);
        const float go = sigmoidf_fast(xproj[xrow + 2 * HSZ] + gbuf[b * 132 + 64 + j]);
        const float cd = tanhf        (xproj[xrow + 3 * HSZ] + gbuf[b * 132 + 96 + j]);
        const int ci = b * HSZ + wg * 32 + j;
        const float cy = gf * cstate[ci] + gi * cd;
        cstate[ci] = cy;
        const float hy = go * tanhf(cy);
        out[(size_t)t * BATCH * HSZ + ci] = hy;
        h_out[ci] = (bf16_t)hy;
    }
}

// ---------------- launch ----------------
extern "C" void kernel_launch(void* const* d_in, const int* in_sizes, int n_in,
                              void* d_out, int out_size, void* d_ws, size_t ws_size,
                              hipStream_t stream) {
    const float* X  = (const float*)d_in[0];   // [256][64][1024]
    const float* Wi = (const float*)d_in[1];   // [4096][1024]
    const float* bi = (const float*)d_in[2];   // [4096]
    const float* Wh = (const float*)d_in[3];   // [4096][1024]
    const float* bh = (const float*)d_in[4];   // [4096]
    float* out = (float*)d_out;                // [256][64][1024]

    // workspace carve-up (all chunks 256B-aligned by construction)
    char* ws = (char*)d_ws;
    size_t off = 0;
    bf16_t* Xb    = (bf16_t*)(ws + off); off += (size_t)T_STEPS * BATCH * INSZ * sizeof(bf16_t);
    bf16_t* Wib   = (bf16_t*)(ws + off); off += (size_t)G4H * INSZ * sizeof(bf16_t);
    bf16_t* Whb   = (bf16_t*)(ws + off); off += (size_t)G4H * HSZ * sizeof(bf16_t);
    float*  xproj = (float*) (ws + off); off += (size_t)T_STEPS * BATCH * G4H * sizeof(float);
    bf16_t* hb0   = (bf16_t*)(ws + off); off += (size_t)BATCH * HSZ * sizeof(bf16_t);
    bf16_t* hb1   = (bf16_t*)(ws + off); off += (size_t)BATCH * HSZ * sizeof(bf16_t);
    float*  cst   = (float*) (ws + off); off += (size_t)BATCH * HSZ * sizeof(float);

    // deterministic init of recurrent state (h0 = c0 = 0)
    hipMemsetAsync(hb0, 0, (size_t)BATCH * HSZ * sizeof(bf16_t), stream);
    hipMemsetAsync(cst, 0, (size_t)BATCH * HSZ * sizeof(float), stream);

    // 1) convert to bf16
    {
        const int nX = T_STEPS * BATCH * INSZ;
        const int nW = G4H * INSZ;
        f32_to_bf16_kernel<<<nX / (8 * 256), 256, 0, stream>>>(X,  Xb,  nX);
        f32_to_bf16_kernel<<<nW / (8 * 256), 256, 0, stream>>>(Wi, Wib, nW);
        f32_to_bf16_kernel<<<nW / (8 * 256), 256, 0, stream>>>(Wh, Whb, nW);
    }

    // 2) bulk GEMM: xproj = X @ Wi^T + bi + bh   (M=16384, N=4096, K=1024)
    {
        dim3 grid(G4H / 128, (T_STEPS * BATCH) / 128);  // 32 x 128
        gemm_xproj_kernel<<<grid, 256, 40960, stream>>>(Xb, Wib, bi, bh, xproj);
    }

    // 3) sequential recurrence: stream order carries the t->t+1 dependency;
    //    ping-pong bf16 h buffers avoid cross-WG RAW hazards within a step.
    for (int t = 0; t < T_STEPS; ++t) {
        const bf16_t* hin  = (t & 1) ? hb1 : hb0;
        bf16_t*       hout = (t & 1) ? hb0 : hb1;
        lstm_step_kernel<<<32, 256, 64512, stream>>>(t, xproj, Whb, hin, hout, cst, out);
    }
}
